// RNNEncoder_65292092833946
// MI455X (gfx1250) — matrix operand order repaired and checked by reading, not compiled, not run on hardware
//
#include <hip/hip_runtime.h>

// ---------------------------------------------------------------------------
// GRU (B=256, T=1024, F=128, H=256) for MI455X / gfx1250.
// bf16 WMMA (v_wmma_f32_16x16x32_bf16) for all matmuls, f32 state/accum.
// Persistent 32-block kernel, one wave per 16x16 tile of h, global atomic
// barrier per timestep. Latency-bound by the serial T recurrence; all
// per-step traffic (h, weights) stays in L2 (192MB), x streamed once as bf16.
// ---------------------------------------------------------------------------

typedef __bf16 bf16;
typedef __attribute__((ext_vector_type(16))) __bf16 v16bf;
typedef __attribute__((ext_vector_type(8)))  __bf16 v8bf;
typedef __attribute__((ext_vector_type(8)))  float  v8f;

#define B_SZ 256
#define T_SZ 1024
#define F_SZ 128
#define H_SZ 256
#define G3   (3 * H_SZ)
#define NBLK 32
#define NTHREADS 256   // 8 wave32 waves per block -> 256 waves = 256 tiles

// Workspace layout (byte offsets, all 256B aligned)
#define WS_CTR   ((size_t)0)
#define WS_XBF   ((size_t)256)
#define XBF_B    ((size_t)B_SZ * T_SZ * F_SZ * 2)            // 64 MiB
#define WS_WIH   (WS_XBF + XBF_B)
#define WIH_B    ((size_t)G3 * F_SZ * 2)
#define WS_WHH   (WS_WIH + WIH_B)
#define WHH_B    ((size_t)G3 * H_SZ * 2)
#define WS_HF32  (WS_WHH + WHH_B)
#define HF32_B   ((size_t)2 * B_SZ * H_SZ * 4)               // double buffer
#define WS_HBF   (WS_HF32 + HF32_B)

// --- WMMA fragment helpers -------------------------------------------------
// A (16-bit 16x32): lane L -> row L&15; elems 0..15 = contiguous K block of 16,
// starting at kb = 16*(L>>4) (matches B).
__device__ __forceinline__ v16bf ldA(const bf16* __restrict__ p) {
  v8bf lo = *reinterpret_cast<const v8bf*>(p);
  v8bf hi = *reinterpret_cast<const v8bf*>(p + 8);
  return __builtin_shufflevector(lo, hi, 0,1,2,3,4,5,6,7,8,9,10,11,12,13,14,15);
}
// B (16-bit 32x16): lane L -> col L&15; elems 0..15 = contiguous K block of 16,
// starting at kb = 16*(L>>4) (this code's order; matches A).
__device__ __forceinline__ v16bf ldB(const bf16* __restrict__ p) {
  v8bf lo = *reinterpret_cast<const v8bf*>(p);
  v8bf hi = *reinterpret_cast<const v8bf*>(p + 8);
  return __builtin_shufflevector(lo, hi, 0,1,2,3,4,5,6,7,8,9,10,11,12,13,14,15);
}
__device__ __forceinline__ v8f wmma_bf16(v16bf a, v16bf b, v8f c) {
  return __builtin_amdgcn_wmma_f32_16x16x32_bf16(
      /*neg_a=*/false, a, /*neg_b=*/false, b,
      /*c_mod=*/(short)0, c, /*reuse_a=*/false, /*reuse_b=*/false);
}
__device__ __forceinline__ float sigmoidf_fast(float v) {
  return 1.0f / (1.0f + __expf(-v));
}

// --- Phase 0: f32 -> bf16 conversion of x / W_ih / W_hh, zero h0 + barrier --
__global__ void gru_prep(const float* __restrict__ x,
                         const float* __restrict__ wih,
                         const float* __restrict__ whh,
                         bf16* __restrict__ xbf,
                         bf16* __restrict__ wihbf,
                         bf16* __restrict__ whhbf,
                         float* __restrict__ hf32,
                         bf16* __restrict__ hbf,
                         unsigned* __restrict__ ctr) {
  size_t i = (size_t)blockIdx.x * blockDim.x + threadIdx.x;
  size_t stride = (size_t)gridDim.x * blockDim.x;
  const size_t XN = (size_t)B_SZ * T_SZ * F_SZ;
  for (size_t j = i; j < XN; j += stride) xbf[j] = (bf16)x[j];
  for (size_t j = i; j < (size_t)G3 * F_SZ; j += stride) wihbf[j] = (bf16)wih[j];
  for (size_t j = i; j < (size_t)G3 * H_SZ; j += stride) whhbf[j] = (bf16)whh[j];
  for (size_t j = i; j < (size_t)B_SZ * H_SZ; j += stride) {   // h0 = 0 (buffer 0)
    hf32[j] = 0.0f;
    hbf[j]  = (bf16)0.0f;
  }
  if (i == 0) *ctr = 0u;
}

// --- Phase 1: persistent recurrent scan ------------------------------------
__global__ __launch_bounds__(NTHREADS, 1)
void gru_scan(const float* __restrict__ b_ih, const float* __restrict__ b_hh,
              const bf16* __restrict__ xbf,
              const bf16* __restrict__ wihbf, const bf16* __restrict__ whhbf,
              float* __restrict__ hf32, bf16* __restrict__ hbf,
              float* __restrict__ out, unsigned* __restrict__ ctr) {
  const int lane = threadIdx.x & 31;
  const int wave = threadIdx.x >> 5;
  const int tile = blockIdx.x * (NTHREADS / 32) + wave;  // 0..255
  const int m = tile >> 4;        // B-tile   (0..15)
  const int n = tile & 15;        // H-tile   (0..15)
  const int lo = lane & 15;
  const int hi = lane >> 4;
  const int col  = n * 16 + lo;   // hidden-unit column j
  const int aRow = m * 16 + lo;   // batch row for A fragments
  const int akA  = hi * 16;       // A-fragment K base
  const int akB  = hi * 16;       // B-fragment K base

  // Per-column biases are loop invariant: hoist out of the T loop.
  const float br  = b_ih[col]          + b_hh[col];
  const float bz  = b_ih[H_SZ + col]   + b_hh[H_SZ + col];
  const float bxn = b_ih[2 * H_SZ + col];
  const float bhn = b_hh[2 * H_SZ + col];

  // Gate weight rows (row-major over output channel -> contiguous K loads).
  const bf16* wr = wihbf + (size_t)(col)            * F_SZ;
  const bf16* wz = wihbf + (size_t)(H_SZ + col)     * F_SZ;
  const bf16* wn = wihbf + (size_t)(2 * H_SZ + col) * F_SZ;
  const bf16* ur = whhbf + (size_t)(col)            * H_SZ;
  const bf16* uz = whhbf + (size_t)(H_SZ + col)     * H_SZ;
  const bf16* un = whhbf + (size_t)(2 * H_SZ + col) * H_SZ;

  for (int t = 0; t < T_SZ; ++t) {
    const int cur = t & 1, nxt = cur ^ 1;
    v8f ar = {}, az = {}, ax = {}, ah = {};   // r, z, gx_n, gh_n accumulators

    // ---- x_t @ W_ih^T  (K = 128, 4 WMMA k-steps, A shared by 3 gates) ----
    const bf16* xp = xbf + (size_t)aRow * (T_SZ * F_SZ) + (size_t)t * F_SZ;
    if (t + 1 < T_SZ) __builtin_prefetch(xp + F_SZ, 0, 0);  // next timestep row
#pragma unroll
    for (int kk = 0; kk < F_SZ; kk += 32) {
      v16bf a = ldA(xp + kk + akA);
      ar = wmma_bf16(a, ldB(wr + kk + akB), ar);
      az = wmma_bf16(a, ldB(wz + kk + akB), az);
      ax = wmma_bf16(a, ldB(wn + kk + akB), ax);
    }

    // ---- h_t @ W_hh^T  (K = 256, 8 WMMA k-steps) -------------------------
    const bf16* hp = hbf + (size_t)cur * (B_SZ * H_SZ) + (size_t)aRow * H_SZ;
#pragma unroll
    for (int kk = 0; kk < H_SZ; kk += 32) {
      v16bf a = ldA(hp + kk + akA);
      ar = wmma_bf16(a, ldB(ur + kk + akB), ar);
      az = wmma_bf16(a, ldB(uz + kk + akB), az);
      ah = wmma_bf16(a, ldB(un + kk + akB), ah);
    }

    // ---- pointwise gate combine + state update (on C-layout) -------------
    const float* hcur = hf32 + (size_t)cur * (B_SZ * H_SZ);
    float*       hnx  = hf32 + (size_t)nxt * (B_SZ * H_SZ);
    bf16*        hbnx = hbf  + (size_t)nxt * (B_SZ * H_SZ);
#pragma unroll
    for (int v = 0; v < 8; ++v) {
      const int row = m * 16 + v + 8 * hi;       // C/D layout row for VGPR v
      const size_t idx = (size_t)row * H_SZ + col;
      float r  = sigmoidf_fast(ar[v] + br);
      float z  = sigmoidf_fast(az[v] + bz);
      float ng = tanhf(ax[v] + bxn + r * (ah[v] + bhn));
      float hold = hcur[idx];
      float hnew = (1.0f - z) * ng + z * hold;
      hnx[idx]  = hnew;
      hbnx[idx] = (bf16)hnew;
      if (t == T_SZ - 1) out[idx] = hnew;        // (1,B,H) row-major
    }

    // ---- grid-wide barrier: monotone counter, 32 arrivals per step -------
    __threadfence();
    __syncthreads();
    if (threadIdx.x == 0) {
      __hip_atomic_fetch_add(ctr, 1u, __ATOMIC_RELEASE, __HIP_MEMORY_SCOPE_AGENT);
      const unsigned target = (unsigned)(t + 1) * NBLK;
      while (__hip_atomic_load(ctr, __ATOMIC_ACQUIRE, __HIP_MEMORY_SCOPE_AGENT) < target)
        __builtin_amdgcn_s_sleep(2);
    }
    __syncthreads();
  }
}

// ---------------------------------------------------------------------------
extern "C" void kernel_launch(void* const* d_in, const int* in_sizes, int n_in,
                              void* d_out, int out_size, void* d_ws, size_t ws_size,
                              hipStream_t stream) {
  (void)in_sizes; (void)n_in; (void)out_size; (void)ws_size;
  const float* x   = (const float*)d_in[0];   // (B,T,F)
  const float* wih = (const float*)d_in[1];   // (3H,F)
  const float* whh = (const float*)d_in[2];   // (3H,H)
  const float* bih = (const float*)d_in[3];   // (3H,)
  const float* bhh = (const float*)d_in[4];   // (3H,)

  char* ws = (char*)d_ws;
  unsigned* ctr  = (unsigned*)(ws + WS_CTR);
  bf16* xbf      = (bf16*)(ws + WS_XBF);
  bf16* wihbf    = (bf16*)(ws + WS_WIH);
  bf16* whhbf    = (bf16*)(ws + WS_WHH);
  float* hf32    = (float*)(ws + WS_HF32);
  bf16* hbf      = (bf16*)(ws + WS_HBF);

  gru_prep<<<2048, 256, 0, stream>>>(x, wih, whh, xbf, wihbf, whhbf, hf32, hbf, ctr);
  gru_scan<<<NBLK, NTHREADS, 0, stream>>>(bih, bhh, xbf, wihbf, whhbf,
                                          hf32, hbf, (float*)d_out, ctr);
}
